// ThresholdedCCQMetric_6751688589542
// MI455X (gfx1250) — compile-verified, matrix operand order checked by reading, not run on hardware
//
#include <hip/hip_runtime.h>
#include <hip/hip_bf16.h>

typedef unsigned long long u64;
typedef __attribute__((ext_vector_type(8))) int v8i;

#define BATCH 16
#define HH 512
#define WW 512
#define WPR 8              // 512 bits = 8 u64 words per row
#define WORDS_PER_IMG (HH * WPR)   // 4096
#define MAX_THIN_ITERS 128

// ---------------------------------------------------------------------------
// bit-sliced helpers
// ---------------------------------------------------------------------------
__device__ __forceinline__ u64 westof(u64 L, u64 C) { return (C << 1) | (L >> 63); }
__device__ __forceinline__ u64 eastof(u64 C, u64 R) { return (C >> 1) | (R << 63); }

// sum of 8 one-bit planes -> 4 bit planes (carry-save adder tree)
__device__ __forceinline__ void bitcount8(u64 a, u64 b, u64 c, u64 d,
                                          u64 e, u64 f, u64 g, u64 h,
                                          u64 &b0, u64 &b1, u64 &b2, u64 &b3) {
  u64 s1 = a ^ b ^ c,   c1 = (a & b) | (c & (a ^ b));
  u64 s2 = d ^ e ^ f,   c2 = (d & e) | (f & (d ^ e));
  u64 s3 = g ^ h ^ s1,  c3 = (g & h) | (s1 & (g ^ h));
  u64 s4 = s2 ^ s3,     c4 = s2 & s3;                 // weight 1
  u64 s5 = c1 ^ c2 ^ c3, c5 = (c1 & c2) | (c3 & (c1 ^ c2));
  u64 s6 = s5 ^ c4,     c6 = s5 & c4;                 // weight 2
  b0 = s4; b1 = s6; b2 = c5 ^ c6; b3 = c5 & c6;       // weights 4, 8
}

// horizontal dilation radius 3 / 2 on a 64-bit row segment with neighbors
__device__ __forceinline__ u64 hdil3(u64 L, u64 C, u64 R) {
  u64 x = C;
  x |= (C << 1) | (L >> 63);  x |= (C >> 1) | (R << 63);
  x |= (C << 2) | (L >> 62);  x |= (C >> 2) | (R << 62);
  x |= (C << 3) | (L >> 61);  x |= (C >> 3) | (R << 61);
  return x;
}
__device__ __forceinline__ u64 hdil2(u64 L, u64 C, u64 R) {
  u64 x = C;
  x |= (C << 1) | (L >> 63);  x |= (C >> 1) | (R << 63);
  x |= (C << 2) | (L >> 62);  x |= (C >> 2) | (R << 62);
  return x;
}

// ---------------------------------------------------------------------------
// Kernel 1: binarize + bit-pack + in-LDS Zhang-Suen thinning to convergence.
// One block per (image, mask) pair; whole packed image (32KB) lives in LDS.
// ---------------------------------------------------------------------------
__global__ __launch_bounds__(1024)
void thin_kernel(const float* __restrict__ y_pred,
                 const float* __restrict__ y_true,
                 u64* __restrict__ pSkel, u64* __restrict__ gSkel) {
  __shared__ u64 img[HH][WPR];       // 32 KB packed image
  __shared__ int s_changed;

  const int tid  = threadIdx.x;
  const int lane = tid & 31;
  const int wv   = tid >> 5;         // 32 waves of wave32
  const bool isPred = (blockIdx.x < BATCH);
  const int b = isPred ? blockIdx.x : blockIdx.x - BATCH;
  const float* src = (isPred ? y_pred : y_true) + (size_t)b * HH * WW;
  u64* dst = (isPred ? pSkel : gSkel) + (size_t)b * WORDS_PER_IMG;

  // --- binarize + pack via wave32 ballot: 32 pixels -> one 32-bit half word
  unsigned int* img32 = (unsigned int*)&img[0][0];
  for (int k = wv; k < HH * WPR * 2; k += 32) {       // 8192 half-words
    float v = src[(size_t)k * 32 + lane];
    u64 m = __ballot(v > 0.5f);                       // wave32: low 32 bits
    if (lane == 0) img32[k] = (unsigned int)m;
  }
  if (tid == 0) s_changed = 0;
  __syncthreads();

  // --- thinning: thread owns 4 consecutive words of one row
  const int r     = tid >> 1;
  const int wbase = (tid & 1) * 4;

  for (int iter = 0; iter < MAX_THIN_ITERS; ++iter) {
    int local = 0;
    for (int sub = 0; sub < 2; ++sub) {
      // read phase: 3 rows x 6 words neighborhood (zero padded)
      u64 nb[3][6];
      #pragma unroll
      for (int dr = 0; dr < 3; ++dr) {
        int rr = r + dr - 1;
        #pragma unroll
        for (int j = 0; j < 6; ++j) {
          int ww = wbase - 1 + j;
          nb[dr][j] = ((unsigned)rr < (unsigned)HH && (unsigned)ww < (unsigned)WPR)
                          ? img[rr][ww] : 0ull;
        }
      }
      __syncthreads();

      u64 neww[4];
      #pragma unroll
      for (int k = 0; k < 4; ++k) {
        u64 uL = nb[0][k], uC = nb[0][k + 1], uR = nb[0][k + 2];
        u64 mL = nb[1][k], mC = nb[1][k + 1], mR = nb[1][k + 2];
        u64 dL = nb[2][k], dC = nb[2][k + 1], dR = nb[2][k + 2];
        u64 P2 = uC;
        u64 P3 = eastof(uC, uR);
        u64 P4 = eastof(mC, mR);
        u64 P5 = eastof(dC, dR);
        u64 P6 = dC;
        u64 P7 = westof(dL, dC);
        u64 P8 = westof(mL, mC);
        u64 P9 = westof(uL, uC);

        u64 b0, b1, b2, b3;
        bitcount8(P2, P3, P4, P5, P6, P7, P8, P9, b0, b1, b2, b3);
        u64 condB = (b1 | b2 | b3) & ~(b3 | (b2 & b1 & b0)); // 2 <= Bn <= 6

        u64 a0, a1, a2, a3;
        bitcount8(~P2 & P3, ~P3 & P4, ~P4 & P5, ~P5 & P6,
                  ~P6 & P7, ~P7 & P8, ~P8 & P9, ~P9 & P2,
                  a0, a1, a2, a3);
        u64 condA = a0 & ~(a1 | a2 | a3);                    // A == 1

        u64 cond = condB & condA;
        cond &= (sub == 0) ? (~(P2 & P4 & P6) & ~(P4 & P6 & P8))
                           : (~(P2 & P4 & P8) & ~(P2 & P6 & P8));
        neww[k] = mC & ~cond;
        local |= (neww[k] != mC) ? 1 : 0;
      }
      #pragma unroll
      for (int k = 0; k < 4; ++k) img[r][wbase + k] = neww[k];
      __syncthreads();
    }
    if (local) atomicOr(&s_changed, 1);
    __syncthreads();
    int done = (s_changed == 0);
    __syncthreads();
    if (tid == 0) s_changed = 0;
    if (done) break;
    __syncthreads();
  }
  __syncthreads();

  // --- store skeleton
  #pragma unroll
  for (int k = 0; k < 4; ++k) dst[r * WPR + wbase + k] = img[r][wbase + k];
}

// ---------------------------------------------------------------------------
// Kernel 2: radius-3 disk dilation (== d^2 <= 9) + per-word popcounts.
// One thread per packed word; emits TP/FP/FN partials as bytes (<= 64 each).
// ---------------------------------------------------------------------------
__global__ __launch_bounds__(256)
void count_kernel(const u64* __restrict__ pSkel, const u64* __restrict__ gSkel,
                  unsigned char* __restrict__ tpB, unsigned char* __restrict__ fpB,
                  unsigned char* __restrict__ fnB) {
  const int idx = blockIdx.x * 256 + threadIdx.x;     // [0, 65536)
  if (idx >= BATCH * WORDS_PER_IMG) return;
  const int b = idx >> 12;
  const int rw = idx & (WORDS_PER_IMG - 1);
  const int r = rw >> 3;
  const int w = rw & 7;
  const u64* g = gSkel + (size_t)b * WORDS_PER_IMG;
  const u64* p = pSkel + (size_t)b * WORDS_PER_IMG;

  auto ld = [](const u64* base, int rr, int ww) -> u64 {
    return ((unsigned)rr < (unsigned)HH && (unsigned)ww < (unsigned)WPR)
               ? base[rr * WPR + ww] : 0ull;
  };

  u64 gdil, pdil;
  {
    // dy = 0: radius 3 ; |dy| = 1,2: radius 2 ; |dy| = 3: radius 0
    gdil  = hdil3(ld(g, r, w - 1),     ld(g, r, w),     ld(g, r, w + 1));
    gdil |= hdil2(ld(g, r - 1, w - 1), ld(g, r - 1, w), ld(g, r - 1, w + 1));
    gdil |= hdil2(ld(g, r + 1, w - 1), ld(g, r + 1, w), ld(g, r + 1, w + 1));
    gdil |= hdil2(ld(g, r - 2, w - 1), ld(g, r - 2, w), ld(g, r - 2, w + 1));
    gdil |= hdil2(ld(g, r + 2, w - 1), ld(g, r + 2, w), ld(g, r + 2, w + 1));
    gdil |= ld(g, r - 3, w) | ld(g, r + 3, w);

    pdil  = hdil3(ld(p, r, w - 1),     ld(p, r, w),     ld(p, r, w + 1));
    pdil |= hdil2(ld(p, r - 1, w - 1), ld(p, r - 1, w), ld(p, r - 1, w + 1));
    pdil |= hdil2(ld(p, r + 1, w - 1), ld(p, r + 1, w), ld(p, r + 1, w + 1));
    pdil |= hdil2(ld(p, r - 2, w - 1), ld(p, r - 2, w), ld(p, r - 2, w + 1));
    pdil |= hdil2(ld(p, r + 2, w - 1), ld(p, r + 2, w), ld(p, r + 2, w + 1));
    pdil |= ld(p, r - 3, w) | ld(p, r + 3, w);
  }
  const u64 pc = ld(p, r, w);
  const u64 gc = ld(g, r, w);

  tpB[idx] = (unsigned char)__popcll(pc & gdil);      // p inside tp_area
  fpB[idx] = (unsigned char)__popcll(pc & ~gdil);     // p outside tp_area
  fnB[idx] = (unsigned char)__popcll(gc & ~pdil);     // g outside pred reach
}

// ---------------------------------------------------------------------------
// Kernel 3: sum 4096 popcount bytes per (batch, stat) with the matrix engine.
// One wave per block (EXEC all-1s); 4x V_WMMA_I32_16X16X64_IU8 with B = ones
// reduces 4096 bytes into a wave of i32 accumulators.
// ---------------------------------------------------------------------------
__global__ __launch_bounds__(32)
void wmma_reduce_kernel(const unsigned char* __restrict__ tpB,
                        const unsigned char* __restrict__ fpB,
                        const unsigned char* __restrict__ fnB,
                        int* __restrict__ stats) {       // stats[3][16]
  const int stat = blockIdx.x >> 4;   // 0..2
  const int b    = blockIdx.x & 15;   // 0..15
  const int lane = threadIdx.x;       // wave32, full wave
  const unsigned char* base =
      (stat == 0 ? tpB : (stat == 1 ? fpB : fnB)) + b * WORDS_PER_IMG;
  const int* base32 = (const int*)base;

  v8i ones;
  #pragma unroll
  for (int j = 0; j < 8; ++j) ones[j] = 0x01010101;
  v8i acc;
  #pragma unroll
  for (int j = 0; j < 8; ++j) acc[j] = 0;

  #pragma unroll
  for (int t = 0; t < 4; ++t) {                // 4 x 1024 bytes = 4096 bytes
    v8i a;
    #pragma unroll
    for (int j = 0; j < 8; ++j) a[j] = base32[t * 256 + lane * 8 + j];
    // D[m][n] = sum_k A[m][k] * 1 + C[m][n] : matrix unit as byte-reducer
    acc = __builtin_amdgcn_wmma_i32_16x16x64_iu8(
        /*sgn_a=*/false, a, /*sgn_b=*/false, ones, acc,
        /*reuse_a=*/false, /*reuse_b=*/false);
  }

  int partial = 0;
  #pragma unroll
  for (int j = 0; j < 8; ++j) partial += acc[j];
  // every row-sum is replicated across the 16 columns: sum all lanes, /16
  #pragma unroll
  for (int m = 16; m >= 1; m >>= 1) partial += __shfl_xor(partial, m);
  if (lane == 0) stats[stat * BATCH + b] = partial >> 4;
}

// ---------------------------------------------------------------------------
// Kernel 4: per-batch ratios, mean over batch -> 3 floats
// ---------------------------------------------------------------------------
__global__ void finalize_kernel(const int* __restrict__ stats, float* __restrict__ out) {
  if (threadIdx.x != 0 || blockIdx.x != 0) return;
  float corr = 0.f, comp = 0.f, qual = 0.f;
  for (int b = 0; b < BATCH; ++b) {
    float TP = (float)stats[0 * BATCH + b];
    float FP = (float)stats[1 * BATCH + b];
    float FN = (float)stats[2 * BATCH + b];
    corr += TP / (TP + FP + 1e-12f);
    comp += TP / (TP + FN + 1e-12f);
    qual += TP / (TP + FP + FN + 1e-12f);
  }
  out[0] = corr / BATCH;
  out[1] = comp / BATCH;
  out[2] = qual / BATCH;
}

// ---------------------------------------------------------------------------
extern "C" void kernel_launch(void* const* d_in, const int* in_sizes, int n_in,
                              void* d_out, int out_size, void* d_ws, size_t ws_size,
                              hipStream_t stream) {
  const float* y_pred = (const float*)d_in[0];
  const float* y_true = (const float*)d_in[1];
  float* out = (float*)d_out;

  char* ws = (char*)d_ws;
  u64* pSkel = (u64*)(ws);                                   // 512 KB
  u64* gSkel = (u64*)(ws + 524288);                          // 512 KB
  unsigned char* tpB = (unsigned char*)(ws + 1048576);       // 64 KB
  unsigned char* fpB = (unsigned char*)(ws + 1114112);       // 64 KB
  unsigned char* fnB = (unsigned char*)(ws + 1179648);       // 64 KB
  int* stats = (int*)(ws + 1245184);                         // 3*16 ints

  thin_kernel<<<2 * BATCH, 1024, 0, stream>>>(y_pred, y_true, pSkel, gSkel);
  count_kernel<<<(BATCH * WORDS_PER_IMG) / 256, 256, 0, stream>>>(
      pSkel, gSkel, tpB, fpB, fnB);
  wmma_reduce_kernel<<<3 * BATCH, 32, 0, stream>>>(tpB, fpB, fnB, stats);
  finalize_kernel<<<1, 32, 0, stream>>>(stats, out);
}
